// Triplane_79164837200478
// MI455X (gfx1250) — compile-verified, matrix operand order checked by reading, not run on hardware
//
#include <hip/hip_runtime.h>
#include <hip/hip_bf16.h>
#include <math.h>

// ---------------------------------------------------------------------------
// Triplane + MLP fused kernel for gfx1250 (MI455X), wave32 + WMMA f16.
//   feats[16x32] -> h1[16x256] -> h2[16x256] -> h3[16x256] -> out[16x2]
// All weights live in LDS as f16 (transposed, k-contiguous) for the whole
// kernel; each wave owns one 16-point tile at a time. n-tile loop unrolled 4x
// so 4 independent B-fragment load + WMMA chains are in flight (we run at
// 1 wave/SIMD due to the 325KB LDS footprint, so ILP must come from within).
// ---------------------------------------------------------------------------

typedef _Float16 h8  __attribute__((ext_vector_type(8)));
typedef _Float16 h16 __attribute__((ext_vector_type(16)));
typedef float    f8  __attribute__((ext_vector_type(8)));

#define N_PTS    1048576
#define FEAT     32
#define HID      256
#define OUT_C    2
#define XRES     512
#define TRES     64
#define WAVES    4
#define BLOCK    (WAVES * 32)
#define ASTRIDE  264   // act row stride in halfs (16B aligned, bank-friendly)
#define W2STRIDE 264   // W2/W3 row stride in halfs
#define W1STRIDE 32    // W1 row stride in halfs
#define NTILES   (N_PTS / 16)

// Bilinear sample one channel of a [C,H,W] plane, align_corners=True clamp.
__device__ __forceinline__ float bsample(const float* __restrict__ plane, int H, int W,
                                         float gx, float gy, int c) {
  float ix = (gx + 1.0f) * 0.5f * (float)(W - 1);
  float iy = (gy + 1.0f) * 0.5f * (float)(H - 1);
  float x0f = floorf(ix), y0f = floorf(iy);
  float wx = ix - x0f, wy = iy - y0f;
  int x0 = (int)x0f; x0 = x0 < 0 ? 0 : (x0 > W - 1 ? W - 1 : x0);
  int x1 = (x0 + 1 > W - 1) ? W - 1 : x0 + 1;
  int y0 = (int)y0f; y0 = y0 < 0 ? 0 : (y0 > H - 1 ? H - 1 : y0);
  int y1 = (y0 + 1 > H - 1) ? H - 1 : y0 + 1;
  const float* b = plane + (size_t)c * H * W;
  float v00 = b[y0 * W + x0], v01 = b[y0 * W + x1];
  float v10 = b[y1 * W + x0], v11 = b[y1 * W + x1];
  return v00 * (1.0f - wx) * (1.0f - wy) + v01 * wx * (1.0f - wy)
       + v10 * (1.0f - wx) * wy          + v11 * wx * wy;
}

// One MLP layer on a 16-point tile via WMMA 16x16x32 f16 -> f32.
// A = act[16 x 32*KSTEPS] (f16, LDS), B = Wt[n][k] (f16, LDS, k-contiguous).
// Writes leaky-relu(D + bias) back into act (A-frags hoisted first, so the
// in-place overwrite is hazard-free; same-wave LDS ops are in-order).
template<int KSTEPS, int WSTRIDE>
__device__ __forceinline__ void mlp_layer(_Float16* act, const _Float16* Wt,
                                          const float* bias, int lane) {
  const int n    = lane & 15;              // A row m / B row n
  const int koff = (lane & 16) ? 8 : 0;    // high half-wave takes K+8 slice
  const int mrow = (lane & 16) ? 8 : 0;    // D rows 8..15 for high half-wave

  h16 afr[KSTEPS];
#pragma unroll
  for (int ks = 0; ks < KSTEPS; ++ks) {
    const _Float16* r = act + n * ASTRIDE + ks * 32 + koff;
    h8 lo = *(const h8*)(r);       // k = ks*32 + koff + 0..7
    h8 hi = *(const h8*)(r + 16);  // k = ks*32 + 16 + koff + 0..7
    afr[ks] = __builtin_shufflevector(lo, hi, 0,1,2,3,4,5,6,7,8,9,10,11,12,13,14,15);
  }
#pragma unroll 4
  for (int nt = 0; nt < 16; ++nt) {
    f8 c = {};
#pragma unroll
    for (int ks = 0; ks < KSTEPS; ++ks) {
      const _Float16* wr = Wt + (nt * 16 + n) * WSTRIDE + ks * 32 + koff;
      h8 blo = *(const h8*)(wr);
      h8 bhi = *(const h8*)(wr + 16);
      h16 bfr = __builtin_shufflevector(blo, bhi, 0,1,2,3,4,5,6,7,8,9,10,11,12,13,14,15);
      c = __builtin_amdgcn_wmma_f32_16x16x32_f16(false, afr[ks], false, bfr,
                                                 (short)0, c, false, false);
    }
    float bn = bias[nt * 16 + n];
#pragma unroll
    for (int r = 0; r < 8; ++r) {
      float v = c[r] + bn;
      // leaky_relu(0.01) without compare/cndmask pressure:
      float h = fmaxf(v, 0.0f) + 0.01f * fminf(v, 0.0f);
      act[(r + mrow) * ASTRIDE + nt * 16 + n] = (_Float16)h;
    }
  }
}

__global__ __launch_bounds__(BLOCK, 1)
void triplane_wmma_kernel(const float* __restrict__ coords,
                          const float* __restrict__ xy, const float* __restrict__ yt,
                          const float* __restrict__ xt,
                          const float* __restrict__ w1, const float* __restrict__ b1,
                          const float* __restrict__ w2, const float* __restrict__ b2,
                          const float* __restrict__ w3, const float* __restrict__ b3,
                          const float* __restrict__ w4, const float* __restrict__ b4,
                          float* __restrict__ out) {
  // LDS-resident weights (f16, transposed) + per-wave activation tiles.
  // Total ~325.6 KB of the 320 KB/WGP LDS.
  __shared__ _Float16 sW1[HID * W1STRIDE];
  __shared__ _Float16 sW2[HID * W2STRIDE];
  __shared__ _Float16 sW3[HID * W2STRIDE];
  __shared__ float    sW4[OUT_C * HID];
  __shared__ float    sB1[HID], sB2[HID], sB3[HID], sB4[OUT_C];
  __shared__ _Float16 sAct[WAVES][16 * ASTRIDE];

  const int tid = threadIdx.x;
  // One-time cooperative weight fill (fp32 -> f16, transpose to [n][k]).
  for (int i = tid; i < FEAT * HID; i += BLOCK) {
    int k = i >> 8, n = i & 255;
    sW1[n * W1STRIDE + k] = (_Float16)w1[i];
  }
  for (int i = tid; i < HID * HID; i += BLOCK) {
    int k = i >> 8, n = i & 255;
    sW2[n * W2STRIDE + k] = (_Float16)w2[i];
  }
  for (int i = tid; i < HID * HID; i += BLOCK) {
    int k = i >> 8, n = i & 255;
    sW3[n * W2STRIDE + k] = (_Float16)w3[i];
  }
  for (int i = tid; i < HID * OUT_C; i += BLOCK) {
    int k = i >> 1, o = i & 1;
    sW4[o * HID + k] = w4[i];
  }
  for (int i = tid; i < HID; i += BLOCK) { sB1[i] = b1[i]; sB2[i] = b2[i]; sB3[i] = b3[i]; }
  if (tid < OUT_C) sB4[tid] = b4[tid];
  __syncthreads();

  const int wv = tid >> 5, lane = tid & 31;
  _Float16* act = sAct[wv];
  const int waveGlobal = blockIdx.x * WAVES + wv;
  const int waveCount  = gridDim.x * WAVES;

  for (int tile = waveGlobal; tile < NTILES; tile += waveCount) {
    // Prefetch next tile's coordinate block into cache (global_prefetch_b8).
    {
      int nxt = tile + waveCount;
      if (nxt < NTILES)
        __builtin_prefetch(coords + (size_t)nxt * 48 + lane, 0, 3);
    }

    // ---- Feature sampling: lane = channel, loop over 16 points ----
    for (int p = 0; p < 16; ++p) {
      const float* cp = coords + (size_t)(tile * 16 + p) * 3;
      float x = cp[0], y = cp[1], tc = cp[2];
      float f = bsample(xy, XRES, XRES, x, y, lane)
              + bsample(yt, TRES, XRES, y, tc, lane)
              + bsample(xt, TRES, XRES, x, tc, lane);
      act[p * ASTRIDE + lane] = (_Float16)f;
    }

    // ---- MLP: 32->256->256->256, WMMA f16 with f32 accumulation ----
    mlp_layer<1, W1STRIDE>(act, sW1, sB1, lane);
    mlp_layer<8, W2STRIDE>(act, sW2, sB2, lane);
    mlp_layer<8, W2STRIDE>(act, sW3, sB3, lane);

    // ---- Layer 4 (256->2) + sigmoid: lane = (point, out-channel) ----
    {
      const int p = lane >> 1, o = lane & 1;
      float acc = sB4[o];
      const _Float16* ar = act + p * ASTRIDE;
      const float* wr = sW4 + o * HID;
#pragma unroll 4
      for (int k = 0; k < HID; k += 8) {
        h8 av = *(const h8*)(ar + k);
#pragma unroll
        for (int j = 0; j < 8; ++j) acc += (float)av[j] * wr[k + j];
      }
      out[(size_t)tile * 32 + lane] = 1.0f / (1.0f + __expf(-acc));
    }
  }
}

extern "C" void kernel_launch(void* const* d_in, const int* in_sizes, int n_in,
                              void* d_out, int out_size, void* d_ws, size_t ws_size,
                              hipStream_t stream) {
  const float* coords = (const float*)d_in[0];
  const float* xy     = (const float*)d_in[1];
  const float* yt     = (const float*)d_in[2];
  const float* xt     = (const float*)d_in[3];
  const float* w1     = (const float*)d_in[4];
  const float* b1     = (const float*)d_in[5];
  const float* w2     = (const float*)d_in[6];
  const float* b2     = (const float*)d_in[7];
  const float* w3     = (const float*)d_in[8];
  const float* b3     = (const float*)d_in[9];
  const float* w4     = (const float*)d_in[10];
  const float* b4     = (const float*)d_in[11];
  (void)in_sizes; (void)n_in; (void)out_size; (void)d_ws; (void)ws_size;

  triplane_wmma_kernel<<<512, BLOCK, 0, stream>>>(
      coords, xy, yt, xt, w1, b1, w2, b2, w3, b3, w4, b4, (float*)d_out);
}
